// GRNN_24429773979648
// MI455X (gfx1250) — matrix operand order repaired and checked by reading, not compiled, not run on hardware
//
#include <hip/hip_runtime.h>
#include <hip/hip_fp16.h>

typedef __attribute__((ext_vector_type(16))) _Float16 v16h;
typedef __attribute__((ext_vector_type(8)))  _Float16 v8h;
typedef __attribute__((ext_vector_type(8)))  float    v8f;
typedef __attribute__((ext_vector_type(4)))  unsigned int v4u;
typedef __attribute__((ext_vector_type(8)))  int      v8i;
typedef __attribute__((ext_vector_type(4)))  int      v4i;

#define NN 8192
#define DD 512
#define OO 512

constexpr float kInv2Sig2 = 1.0f / 2048.0f;   // 1/(2*sigma^2), sigma=32
constexpr float kEps      = 1e-8f;

#if defined(__has_builtin)
#if __has_builtin(__builtin_amdgcn_tensor_load_to_lds)
#define HAVE_TDM 1
#endif
#endif

// ---------------------------------------------------------------------------
// TDM: DMA a 128-row x 32-half tile (row stride `rowlen` halves) from global
// to LDS, packed linearly (row-major, 64 B/row). D# per CDNA5 ISA.
// ---------------------------------------------------------------------------
#ifdef HAVE_TDM
__device__ __forceinline__ void tdm_issue_tile(const _Float16* gsrc, unsigned int lds_off,
                                               unsigned int rowlen) {
    const unsigned long long ga = (unsigned long long)(uintptr_t)gsrc;
    v4u g0;
    g0.x = 1u;                                               // count=1 (valid), user mode
    g0.y = lds_off;                                          // lds_addr (bytes)
    g0.z = (unsigned int)ga;                                 // global_addr[31:0]
    g0.w = (unsigned int)((ga >> 32) & 0x01FFFFFFu) | (2u << 30);  // addr[56:32] | type=2
    v8i g1;
    g1[0] = (int)(1u << 16);                                 // data_size=1 -> 2 bytes
    g1[1] = (int)(rowlen << 16);                             // tensor_dim0[15:0] @ 63:48
    g1[2] = (int)((((unsigned int)NN & 0xFFFFu) << 16) | (rowlen >> 16)); // dim0 hi | dim1 lo
    g1[3] = (int)(32u << 16);                                // tensor_dim1 hi=0 | tile_dim0=32
    g1[4] = 128;                                             // tile_dim1=128, tile_dim2=0
    g1[5] = (int)rowlen;                                     // tensor_dim0_stride lo32
    g1[6] = 0;                                               // stride hi16 | dim1_stride lo16
    g1[7] = 0;
    v4i z4 = {0, 0, 0, 0};
#if __clang_major__ >= 23
    v8i z8 = {0, 0, 0, 0, 0, 0, 0, 0};
    __builtin_amdgcn_tensor_load_to_lds(g0, g1, z4, z4, z8, 0);
#else
    __builtin_amdgcn_tensor_load_to_lds(g0, g1, z4, z4, 0);
#endif
}
#else
// Fallback: cooperative 256-thread copy of the same tile.
__device__ __forceinline__ void coop_load_tile(const _Float16* gsrc, _Float16* dst, int tid,
                                               unsigned int rowlen) {
    for (int i = tid; i < 128 * 4; i += 256) {
        const int row = i >> 2, seg = i & 3;
        *(v8h*)(dst + row * 32 + seg * 8) =
            *(const v8h*)(gsrc + (size_t)row * rowlen + seg * 8);
    }
}
#endif

// ---------------------------------------------------------------------------
// Prep: f16 copies of x (row-major + transposed) and per-row squared norms.
// ---------------------------------------------------------------------------
__global__ void k_prep_x(const float* __restrict__ x, _Float16* __restrict__ xh,
                         _Float16* __restrict__ xhT, float* __restrict__ sq) {
    const int row = blockIdx.x;
    const int t   = threadIdx.x;
    __shared__ float red[256];
    float s = 0.0f;
    for (int c = t; c < DD; c += 256) {
        const float v = x[(size_t)row * DD + c];
        s += v * v;
        const _Float16 h = (_Float16)v;
        xh [(size_t)row * DD + c] = h;
        xhT[(size_t)c * NN + row] = h;
    }
    red[t] = s;
    __syncthreads();
    for (int o = 128; o > 0; o >>= 1) {
        if (t < o) red[t] += red[t + o];
        __syncthreads();
    }
    if (t == 0) sq[row] = red[0];
}

__global__ void k_prep_w(const float* __restrict__ W, _Float16* __restrict__ Wh,
                         float* __restrict__ rowsum) {
    const int i = blockIdx.x * blockDim.x + threadIdx.x;
    if (i < OO * DD) Wh[i] = (_Float16)W[i];
    if (i < NN)      rowsum[i] = 0.0f;
}

// ---------------------------------------------------------------------------
// K1: fused Gram + Gaussian kernel. TDM double-buffers 128x32 A/B K-slabs of
// xh into LDS; each wave runs 8 WMMAs per K-step from ds_load_b128 fragments.
// ---------------------------------------------------------------------------
__global__ __launch_bounds__(256, 2)
void k_gram_exp(const _Float16* __restrict__ xh, const float* __restrict__ sq,
                _Float16* __restrict__ wbuf, float* __restrict__ rowsum) {
    __shared__ alignas(32) _Float16 sA[2][128 * 32];
    __shared__ alignas(32) _Float16 sB[2][128 * 32];

    const int wave  = threadIdx.x >> 5;
    const int lane  = threadIdx.x & 31;
    const int lmod  = lane & 15;
    const int lhi   = lane >> 4;                 // 0 or 1
    const int iBase = blockIdx.y * 128 + wave * 16;
    const int jBase = blockIdx.x * 128;
    const int cA    = lhi * 16;                  // A-frag K offset (contiguous k = 16h + s)

    const _Float16* gA = xh + (size_t)(blockIdx.y * 128) * DD;
    const _Float16* gB = xh + (size_t)jBase * DD;

    const _Float16* aF = &sA[0][(wave * 16 + lmod) * 32];
    const _Float16* bF = &sB[0][lmod * 32 + lhi * 16];

    v8f acc[8] = {};

#ifdef HAVE_TDM
    const unsigned int ldsA0 = (unsigned int)(uintptr_t)&sA[0][0];
    const unsigned int ldsA1 = (unsigned int)(uintptr_t)&sA[1][0];
    const unsigned int ldsB0 = (unsigned int)(uintptr_t)&sB[0][0];
    const unsigned int ldsB1 = (unsigned int)(uintptr_t)&sB[1][0];
    if (wave == 0) {
        tdm_issue_tile(gA, ldsA0, DD);
        tdm_issue_tile(gB, ldsB0, DD);
        __builtin_amdgcn_s_wait_tensorcnt(0);
    }
    __syncthreads();
#pragma unroll
    for (int idx = 0; idx < DD / 32; ++idx) {
        const int cur = idx & 1;
        if (idx + 1 < DD / 32 && wave == 0) {
            const int kk2 = (idx + 1) * 32;
            tdm_issue_tile(gA + kk2, cur ? ldsA0 : ldsA1, DD);
            tdm_issue_tile(gB + kk2, cur ? ldsB0 : ldsB1, DD);
        }
        union { v16h v; v8h h[2]; } a;
        a.h[0] = *(const v8h*)(aF + cur * 128 * 32 + cA);
        a.h[1] = *(const v8h*)(aF + cur * 128 * 32 + cA + 8);
#pragma unroll
        for (int t = 0; t < 8; ++t) {
            const v16h b = *(const v16h*)(bF + cur * 128 * 32 + t * 16 * 32);
            acc[t] = __builtin_amdgcn_wmma_f32_16x16x32_f16(
                false, a.v, false, b, (short)0, acc[t], false, false);
        }
        if (wave == 0) __builtin_amdgcn_s_wait_tensorcnt(0);
        __syncthreads();
    }
#else
    for (int idx = 0; idx < DD / 32; ++idx) {
        __syncthreads();
        coop_load_tile(gA + idx * 32, &sA[0][0], threadIdx.x, DD);
        coop_load_tile(gB + idx * 32, &sB[0][0], threadIdx.x, DD);
        __syncthreads();
        union { v16h v; v8h h[2]; } a;
        a.h[0] = *(const v8h*)(aF + cA);
        a.h[1] = *(const v8h*)(aF + cA + 8);
#pragma unroll
        for (int t = 0; t < 8; ++t) {
            const v16h b = *(const v16h*)(bF + t * 16 * 32);
            acc[t] = __builtin_amdgcn_wmma_f32_16x16x32_f16(
                false, a.v, false, b, (short)0, acc[t], false, false);
        }
    }
#endif

    float si[8], rsum[8];
#pragma unroll
    for (int r = 0; r < 8; ++r) { si[r] = sq[iBase + r + lhi * 8]; rsum[r] = 0.0f; }

#pragma unroll
    for (int t = 0; t < 8; ++t) {
        const int jcol = jBase + 16 * t + lmod;
        const float sj = sq[jcol];
#pragma unroll
        for (int r = 0; r < 8; ++r) {
            const float e = fminf((2.0f * acc[t][r] - si[r] - sj) * kInv2Sig2, 0.0f);
            const float w = __expf(e);
            wbuf[(size_t)(iBase + r + lhi * 8) * NN + jcol] = (_Float16)w;
            rsum[r] += w;
        }
    }
#pragma unroll
    for (int r = 0; r < 8; ++r) {
        float v = rsum[r];
        v += __shfl_xor(v, 8, 32);               // reduce within 16-lane half
        v += __shfl_xor(v, 4, 32);
        v += __shfl_xor(v, 2, 32);
        v += __shfl_xor(v, 1, 32);
        if (lmod == 0) atomicAdd(&rowsum[iBase + r + lhi * 8], v);
    }
}

// ---------------------------------------------------------------------------
// K2: O = diag(1/(rowsum+eps)) * wU @ x   (f16 WMMA, K = N = 8192).
// Same TDM double-buffered pipeline as K1, 256 K-steps (unrolled x2 so the
// ping-pong LDS offsets stay compile-time immediates).
// ---------------------------------------------------------------------------
__global__ __launch_bounds__(256, 2)
void k_nadaraya(const _Float16* __restrict__ wbuf, const _Float16* __restrict__ xhT,
                const float* __restrict__ rowsum, _Float16* __restrict__ Oh) {
    __shared__ alignas(32) _Float16 sA[2][128 * 32];
    __shared__ alignas(32) _Float16 sB[2][128 * 32];

    const int wave  = threadIdx.x >> 5;
    const int lane  = threadIdx.x & 31;
    const int lmod  = lane & 15;
    const int lhi   = lane >> 4;
    const int iBase = blockIdx.y * 128 + wave * 16;
    const int dBase = blockIdx.x * 128;
    const int cA    = lhi * 16;

    const _Float16* gA = wbuf + (size_t)(blockIdx.y * 128) * NN;
    const _Float16* gB = xhT + (size_t)dBase * NN;

    const _Float16* aF = &sA[0][(wave * 16 + lmod) * 32];
    const _Float16* bF = &sB[0][lmod * 32 + lhi * 16];

    v8f acc[8] = {};

#ifdef HAVE_TDM
    const unsigned int ldsA0 = (unsigned int)(uintptr_t)&sA[0][0];
    const unsigned int ldsA1 = (unsigned int)(uintptr_t)&sA[1][0];
    const unsigned int ldsB0 = (unsigned int)(uintptr_t)&sB[0][0];
    const unsigned int ldsB1 = (unsigned int)(uintptr_t)&sB[1][0];
    if (wave == 0) {
        tdm_issue_tile(gA, ldsA0, NN);
        tdm_issue_tile(gB, ldsB0, NN);
        __builtin_amdgcn_s_wait_tensorcnt(0);
    }
    __syncthreads();

    auto kstep = [&](int idx, int cur) {
        if (wave == 0 && idx + 1 < NN / 32) {
            const int kk2 = (idx + 1) * 32;
            tdm_issue_tile(gA + kk2, cur ? ldsA0 : ldsA1, NN);
            tdm_issue_tile(gB + kk2, cur ? ldsB0 : ldsB1, NN);
        }
        union { v16h v; v8h h[2]; } a;
        a.h[0] = *(const v8h*)(aF + cur * 128 * 32 + cA);
        a.h[1] = *(const v8h*)(aF + cur * 128 * 32 + cA + 8);
#pragma unroll
        for (int t = 0; t < 8; ++t) {
            const v16h b = *(const v16h*)(bF + cur * 128 * 32 + t * 16 * 32);
            acc[t] = __builtin_amdgcn_wmma_f32_16x16x32_f16(
                false, a.v, false, b, (short)0, acc[t], false, false);
        }
        if (wave == 0) __builtin_amdgcn_s_wait_tensorcnt(0);
        __syncthreads();
    };
    for (int idx = 0; idx < NN / 32; idx += 2) {
        kstep(idx, 0);
        kstep(idx + 1, 1);
    }
#else
    for (int idx = 0; idx < NN / 32; ++idx) {
        __syncthreads();
        coop_load_tile(gA + idx * 32, &sA[0][0], threadIdx.x, NN);
        coop_load_tile(gB + idx * 32, &sB[0][0], threadIdx.x, NN);
        __syncthreads();
        union { v16h v; v8h h[2]; } a;
        a.h[0] = *(const v8h*)(aF + cA);
        a.h[1] = *(const v8h*)(aF + cA + 8);
#pragma unroll
        for (int t = 0; t < 8; ++t) {
            const v16h b = *(const v16h*)(bF + t * 16 * 32);
            acc[t] = __builtin_amdgcn_wmma_f32_16x16x32_f16(
                false, a.v, false, b, (short)0, acc[t], false, false);
        }
    }
#endif

    float rs[8];
#pragma unroll
    for (int r = 0; r < 8; ++r) rs[r] = 1.0f / (rowsum[iBase + r + lhi * 8] + kEps);
#pragma unroll
    for (int t = 0; t < 8; ++t) {
#pragma unroll
        for (int r = 0; r < 8; ++r) {
            Oh[(size_t)(iBase + r + lhi * 8) * DD + dBase + 16 * t + lmod] =
                (_Float16)(acc[t][r] * rs[r]);
        }
    }
}

// ---------------------------------------------------------------------------
// K3: out = O @ W^T + b  (f32 output). W row-major [O,D] is exactly the
// B-fragment layout for W^T (per-lane contiguous K run along a W row).
// ---------------------------------------------------------------------------
__global__ __launch_bounds__(256, 2)
void k_head(const _Float16* __restrict__ Oh, const _Float16* __restrict__ Wh,
            const float* __restrict__ bias, float* __restrict__ out) {
    const int wave  = threadIdx.x >> 5;
    const int lane  = threadIdx.x & 31;
    const int lmod  = lane & 15;
    const int lhi   = lane >> 4;
    const int iBase = blockIdx.y * 128 + wave * 16;
    const int oBase = blockIdx.x * 128;

    v8f acc[8] = {};
    const _Float16* aptr = Oh + (size_t)(iBase + lmod) * DD + lhi * 16;
    const _Float16* bptr[8];
#pragma unroll
    for (int t = 0; t < 8; ++t)
        bptr[t] = Wh + (size_t)(oBase + 16 * t + lmod) * DD + lhi * 16;

#pragma unroll
    for (int s = 0; s < DD / 32; ++s) {
        union { v16h v; v8h h[2]; } a;
        a.h[0] = *(const v8h*)(aptr + 32 * s);
        a.h[1] = *(const v8h*)(aptr + 32 * s + 8);
#pragma unroll
        for (int t = 0; t < 8; ++t) {
            const v16h b = *(const v16h*)(bptr[t] + 32 * s);
            acc[t] = __builtin_amdgcn_wmma_f32_16x16x32_f16(
                false, a.v, false, b, (short)0, acc[t], false, false);
        }
    }

#pragma unroll
    for (int t = 0; t < 8; ++t) {
        const int ocol = oBase + 16 * t + lmod;
        const float bb = bias[ocol];
#pragma unroll
        for (int r = 0; r < 8; ++r) {
            out[(size_t)(iBase + r + lhi * 8) * OO + ocol] = acc[t][r] + bb;
        }
    }
}

// ---------------------------------------------------------------------------
extern "C" void kernel_launch(void* const* d_in, const int* in_sizes, int n_in,
                              void* d_out, int out_size, void* d_ws, size_t ws_size,
                              hipStream_t stream) {
    const float* x = (const float*)d_in[0];
    const float* W = (const float*)d_in[1];
    const float* b = (const float*)d_in[2];
    float* out = (float*)d_out;

    // Carve workspace (~160 MB): xh, xhT, Wh, sq, rowsum, Oh, wU (128 MB f16).
    char* p = (char*)d_ws;
    auto carve = [&](size_t bytes) -> char* {
        char* q = p;
        p += (bytes + 255) & ~(size_t)255;
        return q;
    };
    _Float16* xh     = (_Float16*)carve((size_t)NN * DD * sizeof(_Float16));
    _Float16* xhT    = (_Float16*)carve((size_t)DD * NN * sizeof(_Float16));
    _Float16* Wh     = (_Float16*)carve((size_t)OO * DD * sizeof(_Float16));
    float*    sq     = (float*)   carve((size_t)NN * sizeof(float));
    float*    rowsum = (float*)   carve((size_t)NN * sizeof(float));
    _Float16* Oh     = (_Float16*)carve((size_t)NN * DD * sizeof(_Float16));
    _Float16* wbuf   = (_Float16*)carve((size_t)NN * NN * sizeof(_Float16));

    k_prep_x<<<NN, 256, 0, stream>>>(x, xh, xhT, sq);
    k_prep_w<<<(OO * DD + 255) / 256, 256, 0, stream>>>(W, Wh, rowsum);
    k_gram_exp<<<dim3(NN / 128, NN / 128), 256, 0, stream>>>(xh, sq, wbuf, rowsum);
    k_nadaraya<<<dim3(DD / 128, NN / 128), 256, 0, stream>>>(wbuf, xhT, rowsum, Oh);
    k_head<<<dim3(OO / 128, NN / 128), 256, 0, stream>>>(Oh, Wh, b, out);
}
